// mLSTMCell_24154896072843
// MI455X (gfx1250) — matrix operand order repaired and checked by reading, not compiled, not run on hardware
//
#include <hip/hip_runtime.h>
#include <hip/hip_bf16.h>
#include <math.h>

#define Bc 2
#define Sc 2048
#define Ec 1024
#define NHc 4
#define DHc 256
#define SQ_TILES (Sc/16)   // 128
#define WAVES 4

typedef __attribute__((ext_vector_type(16))) __bf16 v16bf;
typedef __attribute__((ext_vector_type(8)))  __bf16 v8bf;
typedef __attribute__((ext_vector_type(8)))  float  v8f;
typedef __attribute__((ext_vector_type(4)))  int    v4i_;

static __device__ __forceinline__ __bf16 f2bf(float x) {
  unsigned u = __builtin_bit_cast(unsigned, x);
  unsigned r = u + 0x7FFFu + ((u >> 16) & 1u);   // round-to-nearest-even
  unsigned short hs = (unsigned short)(r >> 16);
  return __builtin_bit_cast(__bf16, hs);
}

static __device__ __forceinline__ unsigned pack2(float lo, float hi) {
  unsigned a = (unsigned)__builtin_bit_cast(unsigned short, f2bf(lo));
  unsigned b = (unsigned)__builtin_bit_cast(unsigned short, f2bf(hi));
  return a | (b << 16);
}

static __device__ __forceinline__ v16bf cat16(v8bf lo, v8bf hi) {
  return __builtin_shufflevector(lo, hi, 0,1,2,3,4,5,6,7,8,9,10,11,12,13,14,15);
}

// ---- async global->LDS staging (CDNA5 ASYNCcnt path), with portable fallback ----
#if defined(__has_builtin) && __has_builtin(__builtin_amdgcn_global_load_async_to_lds_b128)
#define HAVE_ASYNC 1
typedef __attribute__((address_space(1))) v4i_* gp1_t;
typedef __attribute__((address_space(3))) v4i_* lp3_t;
static __device__ __forceinline__ void cp16(__bf16* d, const __bf16* s) {
  __bf16* sn = const_cast<__bf16*>(s);
  __builtin_amdgcn_global_load_async_to_lds_b128((gp1_t)sn, (lp3_t)d, 0, 0);
}
static __device__ __forceinline__ void wait_async() {
  asm volatile("s_wait_asynccnt 0" ::: "memory");
}
#else
#define HAVE_ASYNC 0
static __device__ __forceinline__ void cp16(__bf16* d, const __bf16* s) {
  *(v8bf*)d = *(const v8bf*)s;                  // global_load_b128 + ds_store_b128
}
static __device__ __forceinline__ void wait_async() {}  // ds stores fenced by barrier
#endif

// ---------------- Prep A: q,k -> bf16 head-major [B,NH,S,DH] ----------------
__global__ void cvt_qk_kernel(const float* __restrict__ q, const float* __restrict__ k,
                              __bf16* __restrict__ qbf, __bf16* __restrict__ kbf) {
  const float* src = blockIdx.y ? k : q;
  __bf16* dst = blockIdx.y ? kbf : qbf;
  size_t base = ((size_t)blockIdx.x * blockDim.x + threadIdx.x) * 8;
  size_t bs = base / Ec; int f = (int)(base % Ec);
  int h = f / DHc, d = f % DHc;
  size_t o = ((bs / Sc * NHc + h) * Sc + (bs % Sc)) * DHc + d;
  const float4* p = (const float4*)(src + base);
  float4 x = p[0], y = p[1];
  uint4 r;
  r.x = pack2(x.x, x.y); r.y = pack2(x.z, x.w);
  r.z = pack2(y.x, y.y); r.w = pack2(y.z, y.w);
  *(uint4*)(dst + o) = r;
}

// ---------------- Prep B: v -> bf16 transposed [B,NH,DH,S] ----------------
__global__ void cvt_vt_kernel(const float* __restrict__ v, __bf16* __restrict__ vt) {
  __shared__ unsigned short tb[64][65];
  int bid = blockIdx.x;
  int dt = bid & 3;  bid >>= 2;        // DH/64
  int st = bid & 31; bid >>= 5;        // S/64
  int bh = bid;                        // B*NH
  int b = bh >> 2, h = bh & 3;
  int t = threadIdx.x;
  int col = t & 63, rowq = t >> 6;
  #pragma unroll 4
  for (int i = 0; i < 16; i++) {
    int row = i * 4 + rowq;
    float x = v[((size_t)(b * Sc + st * 64 + row)) * Ec + h * DHc + dt * 64 + col];
    tb[row][col] = __builtin_bit_cast(unsigned short, f2bf(x));
  }
  __syncthreads();
  #pragma unroll 4
  for (int i = 0; i < 16; i++) {
    int row = i * 4 + rowq;
    vt[((size_t)(bh * DHc + dt * 64 + row)) * Sc + st * 64 + col] =
        __builtin_bit_cast(__bf16, tb[col][row]);
  }
}

// ---------------- Pass 1: gate pre-activations (GEMV) ----------------
__global__ void gates_kernel(const float* __restrict__ q, const float* __restrict__ k,
                             const float* __restrict__ v,
                             const float* __restrict__ igw, const float* __restrict__ igb,
                             const float* __restrict__ fgw, const float* __restrict__ fgb,
                             float* __restrict__ igbuf, float* __restrict__ fgbuf) {
  int bs = blockIdx.x;
  int b = bs / Sc, s = bs % Sc;
  int wv = threadIdx.x >> 5;
  int lane = threadIdx.x & 31;
  int h = wv & 3;
  int isF = wv >> 2;
  const float* wr = (isF ? fgw : igw) + h * (3 * Ec);
  int base = (b * Sc + s) * Ec;
  float acc = 0.f;
  for (int f = lane; f < 3 * Ec; f += 32) {
    float x = (f < Ec) ? q[base + f] : (f < 2 * Ec) ? k[base + f - Ec] : v[base + f - 2 * Ec];
    acc += x * wr[f];
  }
  #pragma unroll
  for (int m = 16; m >= 1; m >>= 1) acc += __shfl_xor(acc, m, 32);
  if (lane == 0) {
    float r = acc + (isF ? fgb[h] : igb[h]);
    (isF ? fgbuf : igbuf)[(b * NHc + h) * Sc + s] = r;
  }
}

// ---------------- Pass 2: per-(b,h) scans -> a[], m[], nfl[] ----------------
__global__ void scan_kernel(const float* __restrict__ igbuf, const float* __restrict__ fgbuf,
                            float* __restrict__ abuf, float* __restrict__ mbuf,
                            float* __restrict__ nflbuf) {
  __shared__ float sm[256];
  int bh = blockIdx.x;
  int t = threadIdx.x;
  const float* fgp = fgbuf + bh * Sc;
  const float* igp = igbuf + bh * Sc;
  float lf[8], csl[8], av[8];
  float run = 0.f;
  #pragma unroll
  for (int e = 0; e < 8; e++) {
    float x = fgp[t * 8 + e];
    float ls = (x >= 0.f) ? -log1pf(expf(-x)) : (x - log1pf(expf(x)));  // log_sigmoid
    lf[e] = ls; run += ls; csl[e] = run;
  }
  sm[t] = run; __syncthreads();
  for (int off = 1; off < 256; off <<= 1) {
    float vv = (t >= off) ? sm[t - off] : 0.f;
    __syncthreads();
    sm[t] += vv;
    __syncthreads();
  }
  float excl = sm[t] - run;
  __syncthreads();
  float runm = -INFINITY;
  #pragma unroll
  for (int e = 0; e < 8; e++) {
    float cs = excl + csl[e];
    float csh = cs - lf[e];
    float a = igp[t * 8 + e] - csh;
    av[e] = a; runm = fmaxf(runm, a);
    abuf[bh * Sc + t * 8 + e] = a;
    csl[e] = cs;
  }
  sm[t] = runm; __syncthreads();
  for (int off = 1; off < 256; off <<= 1) {
    float vv = (t >= off) ? sm[t - off] : -INFINITY;
    __syncthreads();
    sm[t] = fmaxf(sm[t], vv);
    __syncthreads();
  }
  float pm = (t == 0) ? -INFINITY : sm[t - 1];
  #pragma unroll
  for (int e = 0; e < 8; e++) {
    pm = fmaxf(pm, av[e]);
    mbuf[bh * Sc + t * 8 + e] = pm;
    nflbuf[bh * Sc + t * 8 + e] = expf(-(csl[e] + pm));
  }
}

// ---------------- Pass 3: tiled causal decay-attention with WMMA ----------------
__global__ void __launch_bounds__(WAVES * 32, 1)
mlstm_kernel(const __bf16* __restrict__ qbf, const __bf16* __restrict__ kbf,
             const __bf16* __restrict__ vt, const float* __restrict__ outw,
             const float* __restrict__ abuf, const float* __restrict__ mbuf,
             const float* __restrict__ nflbuf, float* __restrict__ out) {
  __shared__ __bf16 Kl[2][32][264];                   // 33 KB, padded rows
  __shared__ __bf16 Vl[2][256][40];                   // 40 KB, padded rows
  __shared__ __bf16 Pbuf[WAVES][16][40];              // 5 KB
  int tid = threadIdx.x;
  int wv = tid >> 5;
  int lane = tid & 31;
  int it0 = (blockIdx.x * WAVES) & (SQ_TILES - 1);    // block's 4 consecutive row tiles
  int bh  = (blockIdx.x * WAVES) >> 7;
  int it = it0 + wv;
  int b = bh >> 2, h = bh & 3;

  int half = lane >> 4;
  int ln = lane & 15;
  int koff = half * 8;

  const __bf16* qrow = qbf + ((size_t)bh * Sc + it * 16 + ln) * DHc;
  const __bf16* kpan = kbf + (size_t)bh * Sc * DHc;
  const __bf16* vpan = vt + (size_t)bh * DHc * Sc;
  const float* ab = abuf + bh * Sc;
  const float* mb = mbuf + bh * Sc;
  const float* nb = nflbuf + bh * Sc;

  // cooperative K/V chunk staging: 32 keys x 256 feats each, all 128 threads
  auto stage = [&](int c, int buf) {
    int j0 = c * 32;
    int key = tid >> 2, seg = tid & 3;                // K: 64 feats per (key,seg)
    const __bf16* ks = kpan + (size_t)(j0 + key) * DHc + seg * 64;
    __bf16* kd = &Kl[buf][key][seg * 64];
    #pragma unroll
    for (int i = 0; i < 8; i++) cp16(kd + i * 8, ks + i * 8);
    #pragma unroll
    for (int rr = 0; rr < 2; rr++) {                  // V: 2 feature-rows per thread
      int row = tid * 2 + rr;
      const __bf16* vs = vpan + (size_t)row * Sc + j0;
      __bf16* vd = &Vl[buf][row][0];
      #pragma unroll
      for (int i = 0; i < 4; i++) cp16(vd + i * 8, vs + i * 8);
    }
  };

  // Q A-fragments: 16 rows x 256 features
  v16bf aq[8];
  #pragma unroll
  for (int kc = 0; kc < 8; kc++) {
    aq[kc] = cat16(*(const v8bf*)(qrow + kc * 32 + koff),
                   *(const v8bf*)(qrow + kc * 32 + 16 + koff));
  }
  float mi[8];
  #pragma unroll
  for (int r = 0; r < 8; r++) mi[r] = mb[it * 16 + r + 8 * half];

  v8f acc[16];
  #pragma unroll
  for (int i = 0; i < 16; i++) acc[i] = (v8f)0.f;
  v8f rs = (v8f)0.f;

  v16bf ones;
  #pragma unroll
  for (int e = 0; e < 16; e++) ones[e] = f2bf(1.0f);

  const float invs = 0.0625f;                         // 1/sqrt(DH)

  int ncmax = ((it0 + WAVES - 1) >> 1) + 1;           // block-uniform trip count
  int ncw   = (it >> 1) + 1;                          // this wave's causal extent
  stage(0, 0);
  for (int c = 0; c < ncmax; c++) {
    wait_async();
    __syncthreads();                                  // chunk c resident in Kl/Vl
    if (c + 1 < ncmax) stage(c + 1, (c + 1) & 1);     // prefetch next chunk (other buf)
    if (c < ncw) {
      int buf = c & 1;
      int jt0 = 2 * c;
      // ---- S = Q K^T for two 16-key tiles; scale exp(a[j]-m[i])/sqrt(DH); mask
      #pragma unroll
      for (int tsub = 0; tsub < 2; tsub++) {
        int jt = jt0 + tsub;
        if (jt <= it) {
          v8f s = (v8f)0.f;
          const __bf16* krow = &Kl[buf][tsub * 16 + ln][8 * half];
          #pragma unroll
          for (int kc = 0; kc < 8; kc++) {
            v16bf bk = cat16(*(const v8bf*)(krow + kc * 32),
                             *(const v8bf*)(krow + kc * 32 + 16));
            s = __builtin_amdgcn_wmma_f32_16x16x32_bf16(false, aq[kc], false, bk,
                                                        (short)0, s, false, false);
          }
          float aj = ab[jt * 16 + ln];
          bool diag = (jt == it);
          #pragma unroll
          for (int r = 0; r < 8; r++) {
            int row = r + 8 * half;
            float sc = invs * __expf(aj - mi[r]);     // <= invs, no overflow
            if (diag && (ln > row)) sc = 0.f;         // in-tile causal mask
            Pbuf[wv][row][tsub * 16 + ln] = f2bf(s[r] * sc);
          }
        } else {
          #pragma unroll
          for (int r = 0; r < 8; r++) Pbuf[wv][r + 8 * half][tsub * 16 + ln] = f2bf(0.f);
        }
      }
      // ---- C-layout -> A-layout via per-wave LDS round trip
      asm volatile("s_wait_dscnt 0" ::: "memory");
      v16bf pa = cat16(*(const v8bf*)&Pbuf[wv][ln][koff],
                       *(const v8bf*)&Pbuf[wv][ln][16 + koff]);
      // row sums (robust normalizer) via ones-matrix WMMA
      rs = __builtin_amdgcn_wmma_f32_16x16x32_bf16(false, pa, false, ones,
                                                   (short)0, rs, false, false);
      // ---- h += P @ V  (K = 32 keys, 16 feature tiles) from LDS
      #pragma unroll
      for (int ft = 0; ft < 16; ft++) {
        const __bf16* vr = &Vl[buf][ft * 16 + ln][8 * half];
        v16bf bv = cat16(*(const v8bf*)(vr), *(const v8bf*)(vr + 16));
        acc[ft] = __builtin_amdgcn_wmma_f32_16x16x32_bf16(false, pa, false, bv,
                                                          (short)0, acc[ft], false, false);
      }
    }
  }

  // ---- epilogue: normalize, group-norm over DH, residual scale, store
  float owv[16];
  #pragma unroll
  for (int ft = 0; ft < 16; ft++) owv[ft] = 1.f + outw[h * DHc + ft * 16 + ln];

  #pragma unroll
  for (int r = 0; r < 8; r++) {
    int rowg = it * 16 + r + 8 * half;
    float nrm = fmaxf(fabsf(rs[r]), nb[rowg]) + 1e-8f;
    float inv = 1.f / nrm;
    float s1 = 0.f, s2 = 0.f;
    float hv[16];
    #pragma unroll
    for (int ft = 0; ft < 16; ft++) {
      float x = acc[ft][r] * inv;
      hv[ft] = x; s1 += x; s2 += x * x;
    }
    #pragma unroll
    for (int msk = 8; msk >= 1; msk >>= 1) {
      s1 += __shfl_xor(s1, msk, 32);
      s2 += __shfl_xor(s2, msk, 32);
    }
    float mean = s1 * (1.f / 256.f);
    float var = s2 * (1.f / 256.f) - mean * mean;
    float rstd = rsqrtf(var + 1e-5f);
    float* op = out + (size_t)(b * Sc + rowg) * Ec + h * DHc;
    #pragma unroll
    for (int ft = 0; ft < 16; ft++) {
      int f = ft * 16 + ln;
      op[f] = (hv[ft] - mean) * rstd * owv[ft];
    }
  }
}

extern "C" void kernel_launch(void* const* d_in, const int* in_sizes, int n_in,
                              void* d_out, int out_size, void* d_ws, size_t ws_size,
                              hipStream_t stream) {
  (void)in_sizes; (void)n_in; (void)out_size; (void)ws_size;
  const float* q   = (const float*)d_in[0];
  const float* k   = (const float*)d_in[1];
  const float* v   = (const float*)d_in[2];
  const float* igw = (const float*)d_in[3];
  const float* igb = (const float*)d_in[4];
  const float* fgw = (const float*)d_in[5];
  const float* fgb = (const float*)d_in[6];
  const float* ow  = (const float*)d_in[7];
  float* out = (float*)d_out;
  float* ws = (float*)d_ws;

  const size_t G = (size_t)Bc * NHc * Sc;             // 16384 gate scalars
  const size_t N = (size_t)Bc * NHc * Sc * DHc;       // 4.19M elements per tensor
  float* igbuf = ws;
  float* fgbuf = ws + G;
  float* abuf  = ws + 2 * G;
  float* mbuf  = ws + 3 * G;
  float* nflbuf= ws + 4 * G;
  __bf16* qbf = (__bf16*)(ws + 5 * G);
  __bf16* kbf = qbf + N;
  __bf16* vtb = qbf + 2 * N;

  cvt_qk_kernel<<<dim3((unsigned)(N / 8 / 256), 2), 256, 0, stream>>>(q, k, qbf, kbf);
  cvt_vt_kernel<<<Bc * NHc * (Sc / 64) * (DHc / 64), 256, 0, stream>>>(v, vtb);
  gates_kernel<<<Bc * Sc, 256, 0, stream>>>(q, k, v, igw, igb, fgw, fgb, igbuf, fgbuf);
  scan_kernel<<<Bc * NHc, 256, 0, stream>>>(igbuf, fgbuf, abuf, mbuf, nflbuf);
  mlstm_kernel<<<(Bc * NHc * SQ_TILES) / WAVES, WAVES * 32, 0, stream>>>(
      qbf, kbf, vtb, ow, abuf, mbuf, nflbuf, out);
}
